// OptimizedEncoderLayer_43619687858420
// MI455X (gfx1250) — matrix operand order repaired and checked by reading, not compiled, not run on hardware
//
#include <hip/hip_runtime.h>
#include <hip/hip_bf16.h>
#include <math.h>

// ---------------------------------------------------------------------------
// Types for CDNA5 WMMA (wave32).
// ---------------------------------------------------------------------------
typedef __bf16 bf16;
typedef __attribute__((ext_vector_type(16))) __bf16 v16bf;
typedef __attribute__((ext_vector_type(8)))  __bf16 v8bf;
typedef __attribute__((ext_vector_type(8)))  float  v8f;

#if defined(__gfx1250__) && __has_builtin(__builtin_amdgcn_global_load_async_to_lds_b128) && __has_builtin(__builtin_amdgcn_s_wait_asynccnt)
#define USE_ASYNC_LDS 1
typedef int v4i __attribute__((vector_size(16)));
typedef __attribute__((address_space(1))) v4i* as1_v4i_ptr;   // "int4 __device__ *"
typedef __attribute__((address_space(3))) v4i* as3_v4i_ptr;
#else
#define USE_ASYNC_LDS 0
#endif

__device__ __forceinline__ bf16 f2bf(float f) {
  unsigned u = __builtin_bit_cast(unsigned, f);
  u = (u + 0x7FFFu + ((u >> 16) & 1u)) >> 16;   // round-to-nearest-even
  return __builtin_bit_cast(bf16, (unsigned short)u);
}

// ---------------------------------------------------------------------------
// fp32 -> bf16 elementwise (src activation)
// ---------------------------------------------------------------------------
__global__ void cvt_f32_bf16(const float* __restrict__ in, bf16* __restrict__ out, int n) {
  int i = blockIdx.x * blockDim.x + threadIdx.x;
  if (i < n) out[i] = f2bf(in[i]);
}

// ---------------------------------------------------------------------------
// fp32 [K][N] -> bf16 transposed [N][K]  (weights; makes B staging a pure copy)
// ---------------------------------------------------------------------------
__global__ void cvt_transpose_f32_bf16(const float* __restrict__ in, bf16* __restrict__ out,
                                       int K, int N) {
  int i = blockIdx.x * blockDim.x + threadIdx.x;
  if (i >= K * N) return;
  int n = i % N, k = i / N;                 // coalesced read over n
  out[(size_t)n * K + k] = f2bf(in[i]);
}

// ---------------------------------------------------------------------------
// WMMA GEMM: Out[M,N] = A[M,K](bf16) * Wt[N,K](bf16, pre-transposed) + bias[N]
//   BN : block tile N, WN : waves along N, EPI : 0=f32 out, 1=bf16 GELU out
// 256 threads = 8 waves. Wave tile = 32 x (BN/WN): SM=2 x SN=BN/(WN*16)
// accumulators; A fragments reused across SN B fragments.
// Double-buffered LDS filled by GLOBAL_LOAD_ASYNC_TO_LDS_B128 (ASYNCcnt
// pipeline: stage k+1 stays in flight while stage k computes).
// All problem dims divide tile sizes -> no epilogue guards.
// ---------------------------------------------------------------------------
template<int BN, int WN, int EPI>
__global__ __launch_bounds__(256) void gemm_wmma(
    const bf16* __restrict__ A, const bf16* __restrict__ Wt,
    const float* __restrict__ bias, void* __restrict__ Outp,
    int M, int K, int N)
{
  constexpr int WM  = 8 / WN;           // waves along M
  constexpr int BM  = WM * 32;          // block tile M (wave M-tile = 32)
  constexpr int WNT = BN / WN;          // wave N-tile
  constexpr int SN  = WNT / 16;         // B sub-fragments per wave (2 or 4)
  constexpr int LDK = 40;               // padded K stride (80B: 16B aligned, bank-skewed)
  constexpr int ACH = (BM * 4) / 256;   // A 16B-chunks per thread
  constexpr int BCH = (BN * 4) < 256 ? 1 : (BN * 4) / 256;  // B chunks per thread
  constexpr int CNT = ACH + BCH;        // async ops in flight per stage

  __shared__ __align__(16) bf16 As[2][BM][LDK];
  __shared__ __align__(16) bf16 Bs[2][BN][LDK];

  const int tid   = threadIdx.x;
  const int m0    = blockIdx.x * BM;
  const int n0    = blockIdx.y * BN;
  const int wave  = tid >> 5;
  const int lane  = tid & 31;
  const int waveM = wave % WM;
  const int waveN = wave / WM;
  const int lrow  = lane & 15;
  const int lhalf = lane >> 4;
  const int nk    = K / 32;

  // ---- per-stage tile copy ----------------------------------------------
  auto stage = [&](int kt, int buf) {
    const int k0 = kt * 32;
#if USE_ASYNC_LDS
#pragma unroll
    for (int it = 0; it < ACH; ++it) {
      int e = tid + it * 256;               // chunk id over BM*4
      int m = e >> 2, j = e & 3;            // row, 16B chunk within 64B row
      __builtin_amdgcn_global_load_async_to_lds_b128(
          (as1_v4i_ptr)(A + (size_t)(m0 + m) * K + k0 + j * 8),
          (as3_v4i_ptr)&As[buf][m][j * 8], 0, 0);
    }
#pragma unroll
    for (int it = 0; it < BCH; ++it) {
      int e = (tid + it * 256) & (BN * 4 - 1);  // dup if BN*4 < 256 (uniform ASYNCcnt)
      int nn = e >> 2, j = e & 3;
      __builtin_amdgcn_global_load_async_to_lds_b128(
          (as1_v4i_ptr)(Wt + (size_t)(n0 + nn) * K + k0 + j * 8),
          (as3_v4i_ptr)&Bs[buf][nn][j * 8], 0, 0);
    }
#else
    uint4 ra[ACH], rb[BCH];
#pragma unroll
    for (int it = 0; it < ACH; ++it) {
      int e = tid + it * 256;
      ra[it] = *(const uint4*)(A + (size_t)(m0 + (e >> 2)) * K + k0 + (e & 3) * 8);
    }
#pragma unroll
    for (int it = 0; it < BCH; ++it) {
      int e = (tid + it * 256) & (BN * 4 - 1);
      rb[it] = *(const uint4*)(Wt + (size_t)(n0 + (e >> 2)) * K + k0 + (e & 3) * 8);
    }
#pragma unroll
    for (int it = 0; it < ACH; ++it) {
      int e = tid + it * 256;
      *(uint4*)&As[buf][e >> 2][(e & 3) * 8] = ra[it];
    }
#pragma unroll
    for (int it = 0; it < BCH; ++it) {
      int e = (tid + it * 256) & (BN * 4 - 1);
      *(uint4*)&Bs[buf][e >> 2][(e & 3) * 8] = rb[it];
    }
#endif
  };

  v8f acc[2][SN];
#pragma unroll
  for (int sm = 0; sm < 2; ++sm)
#pragma unroll
    for (int sn = 0; sn < SN; ++sn)
#pragma unroll
      for (int i = 0; i < 8; ++i) acc[sm][sn][i] = 0.0f;

  stage(0, 0);

  for (int kt = 0; kt < nk; ++kt) {
    const int buf = kt & 1;
    if (kt + 1 < nk) {
      stage(kt + 1, buf ^ 1);
#if USE_ASYNC_LDS
      __builtin_amdgcn_s_wait_asynccnt(CNT);  // stage kt landed; kt+1 in flight
#endif
    } else {
#if USE_ASYNC_LDS
      __builtin_amdgcn_s_wait_asynccnt(0);
#endif
    }
    __syncthreads();

    // ---- fragments from LDS (layouts per ISA 7.12.2) ----
    v16bf af[2], bfr[SN];
#pragma unroll
    for (int sm = 0; sm < 2; ++sm) {
      const bf16* p = &As[buf][waveM * 32 + sm * 16 + lrow][0];
      v8bf lo = *(const v8bf*)(p + lhalf * 8);
      v8bf hi = *(const v8bf*)(p + 16 + lhalf * 8);
#pragma unroll
      for (int i = 0; i < 8; ++i) { af[sm][i] = lo[i]; af[sm][i + 8] = hi[i]; }
    }
#pragma unroll
    for (int sn = 0; sn < SN; ++sn) {
      const bf16* p = &Bs[buf][waveN * WNT + sn * 16 + lrow][0];
      v8bf lo = *(const v8bf*)(p + lhalf * 8);
      v8bf hi = *(const v8bf*)(p + 16 + lhalf * 8);
#pragma unroll
      for (int i = 0; i < 8; ++i) { bfr[sn][i] = lo[i]; bfr[sn][i + 8] = hi[i]; }
    }

    // ---- 2 x SN WMMA (A fragment reused across SN) ----
#pragma unroll
    for (int sm = 0; sm < 2; ++sm)
#pragma unroll
      for (int sn = 0; sn < SN; ++sn)
        acc[sm][sn] = __builtin_amdgcn_wmma_f32_16x16x32_bf16(
            false, af[sm], false, bfr[sn], (short)0, acc[sm][sn], false, false);

    __syncthreads();   // buffer safe to overwrite at kt+2
  }

  // ---- epilogue: VGPR r -> row (lane>>4)*8+r, col lane&15 (no guards) ----
#pragma unroll
  for (int sm = 0; sm < 2; ++sm) {
#pragma unroll
    for (int sn = 0; sn < SN; ++sn) {
#pragma unroll
      for (int r = 0; r < 8; ++r) {
        int row = m0 + waveM * 32 + sm * 16 + lhalf * 8 + r;
        int col = n0 + waveN * WNT + sn * 16 + lrow;
        float v = acc[sm][sn][r] + bias[col];
        if constexpr (EPI == 0) {
          ((float*)Outp)[(size_t)row * N + col] = v;
        } else {
          float g = 0.5f * v * (1.0f + erff(v * 0.70710678118654752f)); // exact GELU
          ((bf16*)Outp)[(size_t)row * N + col] = f2bf(g);
        }
      }
    }
  }
}

// ---------------------------------------------------------------------------
// softmax over P=4 contiguous values (in-place)
// ---------------------------------------------------------------------------
__global__ void softmax_p4(float* __restrict__ attn, int ngroups) {
  int g = blockIdx.x * blockDim.x + threadIdx.x;
  if (g >= ngroups) return;
  float* p = attn + (size_t)g * 4;
  float a0 = p[0], a1 = p[1], a2 = p[2], a3 = p[3];
  float m = fmaxf(fmaxf(a0, a1), fmaxf(a2, a3));
  float e0 = expf(a0 - m), e1 = expf(a1 - m), e2 = expf(a2 - m), e3 = expf(a3 - m);
  float r = 1.0f / (e0 + e1 + e2 + e3);
  p[0] = e0 * r; p[1] = e1 * r; p[2] = e2 * r; p[3] = e3 * r;
}

// ---------------------------------------------------------------------------
// Deformable bilinear sampling + point-weighted sum -> bf16 samp.
// One wave per (b, n, head); lanes = 32 channels of dH.
// ---------------------------------------------------------------------------
__global__ __launch_bounds__(256) void deform_sample(
    const float* __restrict__ off, const float* __restrict__ attn,
    const float* __restrict__ val, bf16* __restrict__ samp)
{
  const int gw   = (blockIdx.x * 256 + threadIdx.x) >> 5;  // (b*4096+n)*8+h
  const int lane = threadIdx.x & 31;
  const int h  = gw & 7;
  const int bn = gw >> 3;            // b*4096 + n
  const int n  = bn & 4095;
  const int b  = bn >> 12;
  const int ix = n & 63, iy = n >> 6;
  const float refx = (ix + 0.5f) * (1.0f / 64.0f);
  const float refy = (iy + 0.5f) * (1.0f / 64.0f);

  const float* offp = off  + (size_t)gw * 8;
  const float* attp = attn + (size_t)gw * 4;
  const float* valb = val  + (size_t)b * 4096 * 256 + h * 32 + lane;

  float acc = 0.0f;
#pragma unroll
  for (int p = 0; p < 4; ++p) {
    float gx = (refx + offp[p * 2 + 0] * (1.0f / 64.0f)) * 64.0f - 0.5f;
    float gy = (refy + offp[p * 2 + 1] * (1.0f / 64.0f)) * 64.0f - 0.5f;
    float aw = attp[p];
    int   x0 = (int)floorf(gx), y0 = (int)floorf(gy);
    float wx = gx - (float)x0,  wy = gy - (float)y0;
    float s = 0.0f;
#pragma unroll
    for (int c = 0; c < 4; ++c) {
      int cx = x0 + (c & 1), cy = y0 + (c >> 1);
      float w = ((c & 1) ? wx : 1.0f - wx) * ((c >> 1) ? wy : 1.0f - wy);
      float valid = (cx >= 0 && cx < 64 && cy >= 0 && cy < 64) ? 1.0f : 0.0f;
      int cxx = min(max(cx, 0), 63), cyy = min(max(cy, 0), 63);
      s += (w * valid) * valb[(size_t)(cyy * 64 + cxx) * 256];
    }
    acc += aw * s;
  }
  samp[(size_t)bn * 256 + h * 32 + lane] = f2bf(acc);
}

// ---------------------------------------------------------------------------
// LayerNorm over D=256: out = LN(a+b)*g + be. One wave per row (wave32).
// Optionally also writes a bf16 copy (feeds the next WMMA GEMM).
// ---------------------------------------------------------------------------
__global__ __launch_bounds__(256) void ln_residual(
    const float* __restrict__ a, const float* __restrict__ b,
    const float* __restrict__ g, const float* __restrict__ be,
    float* __restrict__ out, bf16* __restrict__ out_bf, int rows)
{
  int wave = (blockIdx.x * 256 + threadIdx.x) >> 5;
  int lane = threadIdx.x & 31;
  if (wave >= rows) return;
  const size_t base = (size_t)wave * 256;
  float v[8], s = 0.0f, q = 0.0f;
#pragma unroll
  for (int j = 0; j < 8; ++j) {
    float t = a[base + lane + j * 32] + b[base + lane + j * 32];
    v[j] = t; s += t; q += t * t;
  }
#pragma unroll
  for (int o = 16; o > 0; o >>= 1) {
    s += __shfl_xor(s, o, 32);
    q += __shfl_xor(q, o, 32);
  }
  float mean = s * (1.0f / 256.0f);
  float var  = q * (1.0f / 256.0f) - mean * mean;
  float rs   = rsqrtf(var + 1e-5f);
#pragma unroll
  for (int j = 0; j < 8; ++j) {
    int c = lane + j * 32;
    float o = (v[j] - mean) * rs * g[c] + be[c];
    out[base + c] = o;
    if (out_bf) out_bf[base + c] = f2bf(o);
  }
}

// ---------------------------------------------------------------------------
// Host-side orchestration
// ---------------------------------------------------------------------------
extern "C" void kernel_launch(void* const* d_in, const int* in_sizes, int n_in,
                              void* d_out, int out_size, void* d_ws, size_t ws_size,
                              hipStream_t stream) {
  (void)in_sizes; (void)n_in; (void)out_size; (void)ws_size;
  constexpr int B = 8, N = 4096, D = 256, F = 1024, NH = 8;
  constexpr int M = B * N;                 // 32768 tokens

  const float* src    = (const float*)d_in[0];
  const float* W_off  = (const float*)d_in[1];
  const float* b_off  = (const float*)d_in[2];
  const float* W_attn = (const float*)d_in[3];
  const float* b_attn = (const float*)d_in[4];
  const float* W_v    = (const float*)d_in[5];
  const float* b_v    = (const float*)d_in[6];
  const float* W_o    = (const float*)d_in[7];
  const float* b_o    = (const float*)d_in[8];
  const float* g1     = (const float*)d_in[9];
  const float* be1    = (const float*)d_in[10];
  const float* W1     = (const float*)d_in[11];
  const float* b1     = (const float*)d_in[12];
  const float* W2     = (const float*)d_in[13];
  const float* b2     = (const float*)d_in[14];
  const float* g2     = (const float*)d_in[15];
  const float* be2    = (const float*)d_in[16];
  float* out = (float*)d_out;

  // --- workspace carve-out (256B aligned) --------------------------------
  char* ws = (char*)d_ws;
  auto take = [&](size_t bytes) { char* p = ws; ws += (bytes + 255) & ~(size_t)255; return p; };
  bf16* wb_off  = (bf16*)take((size_t)D * 64 * 2);    // transposed [N][K]
  bf16* wb_attn = (bf16*)take((size_t)D * 32 * 2);
  bf16* wb_v    = (bf16*)take((size_t)D * D * 2);
  bf16* wb_o    = (bf16*)take((size_t)D * D * 2);
  bf16* wb_1    = (bf16*)take((size_t)D * F * 2);
  bf16* wb_2    = (bf16*)take((size_t)F * D * 2);
  bf16* src_bf  = (bf16*)take((size_t)M * D * 2);     // 16MB
  // union region: {off, attn, val, samp} all dead before h is written
  char* uni     = take((size_t)M * F * 2);            // 64MB = h region
  float* off_f  = (float*)uni;                        // 8MB
  float* attn_f = (float*)(uni + (size_t)M * 64 * 4); // 4MB
  float* val_f  = (float*)(uni + (size_t)M * 96 * 4); // 32MB
  bf16*  samp_bf= (bf16*)(uni + (size_t)M * 224 * 4); // 16MB (inside val_f; ends at 44MB < 64MB)
  bf16*  h_bf   = (bf16*)uni;
  float* a_f    = (float*)take((size_t)M * D * 4);    // 32MB, reused as y2
  float* x_f    = (float*)take((size_t)M * D * 4);    // 32MB
  bf16*  x_bf   = (bf16*)take((size_t)M * D * 2);     // 16MB
  float* y2_f   = a_f;

  // 1) convert: src -> bf16 ; weights -> bf16 transposed [N][K]
  cvt_f32_bf16<<<(M * D + 255) / 256, 256, 0, stream>>>(src, src_bf, M * D);
  auto cvtT = [&](const float* in, bf16* o, int K, int Nw) {
    cvt_transpose_f32_bf16<<<(K * Nw + 255) / 256, 256, 0, stream>>>(in, o, K, Nw);
  };
  cvtT(W_off,  wb_off,  D, 64);
  cvtT(W_attn, wb_attn, D, 32);
  cvtT(W_v,    wb_v,    D, D);
  cvtT(W_o,    wb_o,    D, D);
  cvtT(W1,     wb_1,    D, F);
  cvtT(W2,     wb_2,    F, D);

  // 2) projections (WMMA GEMMs, bf16 operands, f32 accumulate)
  gemm_wmma<64, 2, 0><<<dim3(M / 128, 1), 256, 0, stream>>>(src_bf, wb_off,  b_off,  off_f,  M, D, 64);
  gemm_wmma<32, 1, 0><<<dim3(M / 256, 1), 256, 0, stream>>>(src_bf, wb_attn, b_attn, attn_f, M, D, 32);
  softmax_p4<<<(M * NH + 255) / 256, 256, 0, stream>>>(attn_f, M * NH);
  gemm_wmma<128, 2, 0><<<dim3(M / 128, D / 128), 256, 0, stream>>>(src_bf, wb_v, b_v, val_f, M, D, D);

  // 3) deformable bilinear sampling (one wave per (b,n,head)) -> bf16
  deform_sample<<<(M * NH) / 8, 256, 0, stream>>>(off_f, attn_f, val_f, samp_bf);

  // 4) output projection + LN1 (LN1 also emits bf16 x for the MLP GEMM)
  gemm_wmma<128, 2, 0><<<dim3(M / 128, D / 128), 256, 0, stream>>>(samp_bf, wb_o, b_o, a_f, M, D, D);
  ln_residual<<<M / 8, 256, 0, stream>>>(src, a_f, g1, be1, x_f, x_bf, M);

  // 5) MLP: GEMM1 fuses exact GELU -> bf16 h (overlays union); GEMM2 -> y2
  gemm_wmma<128, 2, 1><<<dim3(M / 128, F / 128), 256, 0, stream>>>(x_bf, wb_1, b1, h_bf, M, D, F);
  gemm_wmma<128, 2, 0><<<dim3(M / 128, D / 128), 256, 0, stream>>>(h_bf, wb_2, b2, y2_f, M, F, D);

  // 6) LN2 -> final output
  ln_residual<<<M / 8, 256, 0, stream>>>(x_f, y2_f, g2, be2, out, (bf16*)nullptr, M);
}
